// BiGRU_GCN_67370857005464
// MI455X (gfx1250) — matrix operand order repaired and checked by reading, not compiled, not run on hardware
//
#include <hip/hip_runtime.h>
#include <hip/hip_bf16.h>

typedef __bf16 bf16_t;
typedef __bf16 v16bf __attribute__((ext_vector_type(16)));
typedef __bf16 v8bf  __attribute__((ext_vector_type(8)));
typedef float  v8f   __attribute__((ext_vector_type(8)));

// ---- bf16 weight arena layout inside d_ws (element offsets) ----
enum : int {
  OFF_WF1   = 0,            // w_ih_f1  [384][128]
  OFF_WB1   = 49152,        // w_ih_b1  [384][128]
  OFF_WF2   = 98304,        // w_ih_f2  [384][256]
  OFF_WB2   = 196608,       // w_ih_b2  [384][256]
  OFF_WG1T  = 294912,       // w_g1^T   [128][256]
  OFF_WG2T  = 327680,       // w_g2^T   [ 64][128]
  OFF_WFCT  = 335872,       // w_fc^T   [ 16][ 64] (cols>=10 zero-padded)
  OFF_TOTAL = 336896        // xw1' (transposed [128][N] bf16) starts here
};

static __device__ __forceinline__ v8f wmma_bf16(v16bf a, v16bf b, v8f c) {
  return __builtin_amdgcn_wmma_f32_16x16x32_bf16(false, a, false, b, (short)0, c,
                                                 false, false);
}

static __device__ __forceinline__ v16bf cat16(v8bf lo, v8bf hi) {
  v16bf r;
#pragma unroll
  for (int e = 0; e < 8; ++e) { r[e] = lo[e]; r[8 + e] = hi[e]; }
  return r;
}

// A-fragment (16x32 bf16, M=row) from LDS row-major tile.
static __device__ __forceinline__ v16bf afrag_lds(const bf16_t* base, int stride,
                                                  int lane) {
  int row = lane & 15;
  int kb  = (lane >> 4) * 8;
  const bf16_t* p = base + row * stride + kb;
  v8bf lo = *(const v8bf*)(p);
  v8bf hi = *(const v8bf*)(p + 16);
  return cat16(lo, hi);
}

// B-fragment (32x16 bf16, K x N) from global weights stored [Ncols][Kdim] row-major.
static __device__ __forceinline__ v16bf bfrag_g(const bf16_t* __restrict__ w,
                                                int kdim, int n0, int k0, int lane) {
  int col = n0 + (lane & 15);
  int kb  = k0 + ((lane >> 4) << 3);
  const bf16_t* p = w + (size_t)col * kdim + kb;
  v8bf lo = *(const v8bf*)(p);
  v8bf hi = *(const v8bf*)(p + 16);
  return cat16(lo, hi);
}

// native gfx1250 transcendental tanh; trailing v_nop covers the TRANS
// result-use hazard (ISA 7.4.3) since the compiler can't see inside the asm.
static __device__ __forceinline__ float tanh_hw(float x) {
  float r;
  asm volatile("v_tanh_f32 %0, %1\n\tv_nop" : "=v"(r) : "v"(x));
  return r;
}

static __device__ __forceinline__ float dinv_f(int i, int n) {
  // chain graph + self loop: deg = 2 at ends, 3 interior
  return (i <= 0 || i >= n - 1) ? 0.70710678118654752f : 0.57735026918962576f;
}
static __device__ __forceinline__ unsigned pk2(float a, float b) {
  union { bf16_t h[2]; unsigned u; } t;
  t.h[0] = (bf16_t)a; t.h[1] = (bf16_t)b;
  return t.u;
}

// GRU gates on three 16x16 f32 accumulators -> bf16 h into LDS.
// sigmoid(x) = 0.5 + 0.5*tanh(x/2); (1-z) = 0.5 - 0.5*tanh(zin/2).
// 3 native tanh per element, zero divisions.
static __device__ __forceinline__ void gru_store(v8f aR, v8f aZ, v8f aN,
                                                 const float* __restrict__ bih,
                                                 const float* __restrict__ bhh,
                                                 int g, int lane, bf16_t* dst,
                                                 int dstStride, int sRow) {
  int c = 16 * g + (lane & 15);
  float bir = bih[c], biz = bih[128 + c], bin = bih[256 + c];
  float hr  = bhh[c], hz  = bhh[128 + c], hn  = bhh[256 + c];
  int rb = sRow + 8 * (lane >> 4);
#pragma unroll
  for (int r = 0; r < 8; ++r) {
    float tr = tanh_hw(0.5f * (aR[r] + bir + hr));
    float rg = 0.5f + 0.5f * tr;
    float tz = tanh_hw(0.5f * (aZ[r] + biz + hz));
    float omz = 0.5f - 0.5f * tz;                 // 1 - z
    float ng = tanh_hw(aN[r] + bin + rg * hn);
    dst[(rb + r) * dstStride + c] = (bf16_t)(omz * ng);
  }
}

// ---------------------------------------------------------------------------
// Kernel 0: fp32 -> bf16 weight repack (one tiny launch, ~337K elements)
// ---------------------------------------------------------------------------
__global__ void convert_weights_kernel(const float* __restrict__ wf1,
                                       const float* __restrict__ wb1,
                                       const float* __restrict__ wf2,
                                       const float* __restrict__ wb2,
                                       const float* __restrict__ wg1,
                                       const float* __restrict__ wg2,
                                       const float* __restrict__ wfc,
                                       bf16_t* __restrict__ wsw) {
  int i = blockIdx.x * 256 + threadIdx.x;
  if (i >= OFF_TOTAL) return;
  float v;
  if (i < OFF_WB1)        v = wf1[i];
  else if (i < OFF_WF2)   v = wb1[i - OFF_WB1];
  else if (i < OFF_WB2)   v = wf2[i - OFF_WF2];
  else if (i < OFF_WG1T)  v = wb2[i - OFF_WB2];
  else if (i < OFF_WG2T) { int j = i - OFF_WG1T; int o = j >> 8; int k = j & 255;
                           v = wg1[k * 128 + o]; }
  else if (i < OFF_WFCT) { int j = i - OFF_WG2T; int o = j >> 7; int k = j & 127;
                           v = wg2[k * 64 + o]; }
  else                   { int j = i - OFF_WFCT; int o = j >> 6; int k = j & 63;
                           v = (o < 10) ? wfc[k * 10 + o] : 0.0f; }
  wsw[i] = (bf16_t)v;
}

// ---------------------------------------------------------------------------
// Kernel 1: fused bi-GRU x2 + (h2 @ w_g1), dinv-prescaled, stored transposed.
// 128 threads = 4 waves; each wave owns TWO 16-row strips (S=2) so every
// B-fragment load feeds 2 WMMAs (halves weight-stream bandwidth, the binding
// resource). 128 rows/block, 128 KB LDS, zero barriers (wave-private rows).
// ---------------------------------------------------------------------------
__global__ __launch_bounds__(128) void fused_gru_kernel(
    const float* __restrict__ x, const bf16_t* __restrict__ wsw,
    bf16_t* __restrict__ xw1t,
    const float* __restrict__ bih_f1, const float* __restrict__ bhh_f1,
    const float* __restrict__ bih_b1, const float* __restrict__ bhh_b1,
    const float* __restrict__ bih_f2, const float* __restrict__ bhh_f2,
    const float* __restrict__ bih_b2, const float* __restrict__ bhh_b2, int n) {
  __shared__ __align__(16) bf16_t sX[128 * 128];    // x tile, later h2_fwd
  __shared__ __align__(16) bf16_t sH1[128 * 256];   // h1 (fwd|bwd)
  __shared__ __align__(16) bf16_t sH2b[128 * 128];  // h2_bwd

  int tid = threadIdx.x;
  int lane = tid & 31;
  int wid = tid >> 5;
  int m0 = blockIdx.x * 128;
  int sRow[2] = { (wid * 2 + 0) * 16, (wid * 2 + 1) * 16 };

  // ---- stage x strips: fp32 global -> bf16 LDS (wave-private rows) ----
#pragma unroll
  for (int s = 0; s < 2; ++s) {
    int rr = lane >> 1;
    int cb = (lane & 1) * 64;
    const float2* xp =
        (const float2*)(x + (size_t)(m0 + sRow[s] + rr) * 128 + cb);
    bf16_t* dst = sX + (sRow[s] + rr) * 128 + cb;
#pragma unroll
    for (int j = 0; j < 32; ++j) {
      float2 v = xp[j];
      dst[2 * j]     = (bf16_t)v.x;
      dst[2 * j + 1] = (bf16_t)v.y;
    }
  }

  // ---- layer 1 (K=128): A-frags preloaded, B reused across both strips ----
  v16bf af1[2][4];
#pragma unroll
  for (int s = 0; s < 2; ++s)
#pragma unroll
    for (int kk = 0; kk < 4; ++kk)
      af1[s][kk] = afrag_lds(sX + sRow[s] * 128 + 32 * kk, 128, lane);

  for (int d = 0; d < 2; ++d) {
    const bf16_t* wf = wsw + (d ? OFF_WB1 : OFF_WF1);
    const float* bih = d ? bih_b1 : bih_f1;
    const float* bhh = d ? bhh_b1 : bhh_f1;
    int dOff = d * 128;
    for (int g = 0; g < 8; ++g) {
      v8f aR[2] = {{}, {}}, aZ[2] = {{}, {}}, aN[2] = {{}, {}};
#pragma unroll
      for (int kk = 0; kk < 4; ++kk) {
        v16bf bR = bfrag_g(wf, 128,       16 * g, 32 * kk, lane);
        v16bf bZ = bfrag_g(wf, 128, 128 + 16 * g, 32 * kk, lane);
        v16bf bN = bfrag_g(wf, 128, 256 + 16 * g, 32 * kk, lane);
#pragma unroll
        for (int s = 0; s < 2; ++s) {
          aR[s] = wmma_bf16(af1[s][kk], bR, aR[s]);
          aZ[s] = wmma_bf16(af1[s][kk], bZ, aZ[s]);
          aN[s] = wmma_bf16(af1[s][kk], bN, aN[s]);
        }
      }
#pragma unroll
      for (int s = 0; s < 2; ++s)
        gru_store(aR[s], aZ[s], aN[s], bih, bhh, g, lane, sH1 + dOff, 256,
                  sRow[s]);
    }
  }

  // ---- layer 2 (K=256): A-frags streamed from LDS, B reused across strips ----
  for (int d = 0; d < 2; ++d) {
    const bf16_t* wf = wsw + (d ? OFF_WB2 : OFF_WF2);
    const float* bih = d ? bih_b2 : bih_f2;
    const float* bhh = d ? bhh_b2 : bhh_f2;
    bf16_t* dst = d ? sH2b : sX;  // h2_fwd -> sX (x dead), h2_bwd -> sH2b
    for (int g = 0; g < 8; ++g) {
      v8f aR[2] = {{}, {}}, aZ[2] = {{}, {}}, aN[2] = {{}, {}};
#pragma unroll
      for (int kk = 0; kk < 8; ++kk) {
        v16bf bR = bfrag_g(wf, 256,       16 * g, 32 * kk, lane);
        v16bf bZ = bfrag_g(wf, 256, 128 + 16 * g, 32 * kk, lane);
        v16bf bN = bfrag_g(wf, 256, 256 + 16 * g, 32 * kk, lane);
#pragma unroll
        for (int s = 0; s < 2; ++s) {
          v16bf a = afrag_lds(sH1 + sRow[s] * 256 + 32 * kk, 256, lane);
          aR[s] = wmma_bf16(a, bR, aR[s]);
          aZ[s] = wmma_bf16(a, bZ, aZ[s]);
          aN[s] = wmma_bf16(a, bN, aN[s]);
        }
      }
#pragma unroll
      for (int s = 0; s < 2; ++s)
        gru_store(aR[s], aZ[s], aN[s], bih, bhh, g, lane, dst, 128, sRow[s]);
    }
  }

  // ---- xw1' = dinv(row) * (h2 @ w_g1) -> transposed bf16 to workspace ----
  const bf16_t* wg1 = wsw + OFF_WG1T;
  for (int t = 0; t < 8; ++t) {
    v8f acc[2] = {{}, {}};
#pragma unroll
    for (int kk = 0; kk < 8; ++kk) {
      v16bf b = bfrag_g(wg1, 256, 16 * t, 32 * kk, lane);
#pragma unroll
      for (int s = 0; s < 2; ++s) {
        v16bf a = (kk < 4)
                      ? afrag_lds(sX + sRow[s] * 128 + 32 * kk, 128, lane)
                      : afrag_lds(sH2b + sRow[s] * 128 + 32 * (kk - 4), 128, lane);
        acc[s] = wmma_bf16(a, b, acc[s]);
      }
    }
#pragma unroll
    for (int s = 0; s < 2; ++s) {
      int colg = 16 * t + (lane & 15);
      int rb = m0 + sRow[s] + 8 * (lane >> 4);
      uint4 u4;
      u4.x = pk2(dinv_f(rb + 0, n) * acc[s][0], dinv_f(rb + 1, n) * acc[s][1]);
      u4.y = pk2(dinv_f(rb + 2, n) * acc[s][2], dinv_f(rb + 3, n) * acc[s][3]);
      u4.z = pk2(dinv_f(rb + 4, n) * acc[s][4], dinv_f(rb + 5, n) * acc[s][5]);
      u4.w = pk2(dinv_f(rb + 6, n) * acc[s][6], dinv_f(rb + 7, n) * acc[s][7]);
      *(uint4*)(xw1t + (size_t)colg * n + rb) = u4;
    }
  }
}

// ---------------------------------------------------------------------------
// Kernel 2: chain-GCN stencil -> GEMM -> stencil -> FC, 64 output rows/block.
// ---------------------------------------------------------------------------
__global__ __launch_bounds__(128) void fused_gcn_kernel(
    const bf16_t* __restrict__ wsw, const bf16_t* __restrict__ xw1t,
    const float* __restrict__ bg1, const float* __restrict__ bg2,
    const float* __restrict__ bfc, float* __restrict__ out, int n) {
  // sXW (col-major halo region, 128 cols x 112 rows) is dead after stencil1;
  // its storage is reused for xw2 [96][64] and g2 [64][64].
  __shared__ __align__(16) bf16_t sXW[128 * 112];  // 28672 B
  __shared__ __align__(16) bf16_t sG1[96 * 128];   // 24576 B
  bf16_t* sXW2 = sXW;            // 96*64 elems, alias (valid after barrier)
  bf16_t* sG2  = sXW + 96 * 64;  // 64*64 elems, alias

  int tid = threadIdx.x;
  int lane = tid & 31;
  int wid = tid >> 5;
  int m0 = blockIdx.x * 64;
  int rlo = m0 - 24;  // 8-aligned halo start; rows [m0-24, m0+88)

  // stage 1: load xw1' halo region (zero-pad out-of-range 8-row chunks)
  for (int idx = tid; idx < 128 * 14; idx += 128) {
    int c = idx & 127;
    int j = idx >> 7;
    int gr0 = rlo + j * 8;
    uint4 v;
    if (gr0 >= 0 && gr0 + 8 <= n)
      v = *(const uint4*)(xw1t + (size_t)c * n + gr0);
    else { v.x = 0u; v.y = 0u; v.z = 0u; v.w = 0u; }
    *(uint4*)(sXW + c * 112 + j * 8) = v;
  }
  __syncthreads();

  // stage 2: g1 = dinv(i)*(xw'[i-1]+xw'[i]+xw'[i+1]) + b_g1, rows [m0-16,m0+80)
  {
    int c = tid;  // one column per thread
    float bb = bg1[c];
    for (int rr = 0; rr < 96; ++rr) {
      int gr = m0 - 16 + rr;
      const bf16_t* p = sXW + c * 112 + rr + 7;
      float s = (float)p[0] + (float)p[1] + (float)p[2];
      sG1[rr * 128 + c] = (bf16_t)(dinv_f(gr, n) * s + bb);
    }
  }
  __syncthreads();

  // stage 3: xw2' = dinv(row) * (g1 @ w_g2) -> sXW2 [96][64]
  const bf16_t* wg2 = wsw + OFF_WG2T;
  for (int st = wid; st < 6; st += 4) {
    int lr = st * 16;
    v16bf a4[4];
#pragma unroll
    for (int kk = 0; kk < 4; ++kk)
      a4[kk] = afrag_lds(sG1 + lr * 128 + 32 * kk, 128, lane);
#pragma unroll
    for (int t = 0; t < 4; ++t) {
      v8f acc = {};
#pragma unroll
      for (int kk = 0; kk < 4; ++kk)
        acc = wmma_bf16(a4[kk], bfrag_g(wg2, 128, 16 * t, 32 * kk, lane), acc);
      int c = 16 * t + (lane & 15);
      int rb = lr + 8 * (lane >> 4);
#pragma unroll
      for (int r = 0; r < 8; ++r) {
        int gr = m0 - 16 + rb + r;
        float v = (gr >= 0 && gr < n) ? dinv_f(gr, n) * acc[r] : 0.0f;
        sXW2[(rb + r) * 64 + c] = (bf16_t)v;
      }
    }
  }
  __syncthreads();

  // stage 4: g2 = dinv(i)*(xw2'[i-1]+xw2'[i]+xw2'[i+1]) + b_g2, rows [m0,m0+64)
  {
    int c = tid & 63;
    int half = tid >> 6;
    float bb = bg2[c];
    for (int rr = half * 32; rr < half * 32 + 32; ++rr) {
      int gr = m0 + rr;
      int lr = rr + 16;
      float s = (float)sXW2[lr * 64 + c];
      if (gr - 1 >= 0) s += (float)sXW2[(lr - 1) * 64 + c];
      if (gr + 1 < n)  s += (float)sXW2[(lr + 1) * 64 + c];
      sG2[rr * 64 + c] = (bf16_t)(dinv_f(gr, n) * s + bb);
    }
  }
  __syncthreads();

  // stage 5: out = g2 @ w_fc^T + b_fc (10 valid cols of the 16-col tile)
  const bf16_t* wfc = wsw + OFF_WFCT;
  {
    int lr = wid * 16;
    v16bf a0 = afrag_lds(sG2 + lr * 64,      64, lane);
    v16bf a1 = afrag_lds(sG2 + lr * 64 + 32, 64, lane);
    v8f acc = {};
    acc = wmma_bf16(a0, bfrag_g(wfc, 64, 0, 0,  lane), acc);
    acc = wmma_bf16(a1, bfrag_g(wfc, 64, 0, 32, lane), acc);
    int nn = lane & 15;
    if (nn < 10) {
      float bb = bfc[nn];
      int rb = m0 + lr + 8 * (lane >> 4);
#pragma unroll
      for (int r = 0; r < 8; ++r)
        out[(size_t)(rb + r) * 10 + nn] = acc[r] + bb;
    }
  }
}

// ---------------------------------------------------------------------------
extern "C" void kernel_launch(void* const* d_in, const int* in_sizes, int n_in,
                              void* d_out, int out_size, void* d_ws, size_t ws_size,
                              hipStream_t stream) {
  const float* x       = (const float*)d_in[0];
  const float* w_ih_f1 = (const float*)d_in[1];
  const float* b_ih_f1 = (const float*)d_in[2];
  const float* b_hh_f1 = (const float*)d_in[3];
  const float* w_ih_b1 = (const float*)d_in[4];
  const float* b_ih_b1 = (const float*)d_in[5];
  const float* b_hh_b1 = (const float*)d_in[6];
  const float* w_ih_f2 = (const float*)d_in[7];
  const float* b_ih_f2 = (const float*)d_in[8];
  const float* b_hh_f2 = (const float*)d_in[9];
  const float* w_ih_b2 = (const float*)d_in[10];
  const float* b_ih_b2 = (const float*)d_in[11];
  const float* b_hh_b2 = (const float*)d_in[12];
  const float* w_g1    = (const float*)d_in[13];
  const float* b_g1    = (const float*)d_in[14];
  const float* w_g2    = (const float*)d_in[15];
  const float* b_g2    = (const float*)d_in[16];
  const float* w_fc    = (const float*)d_in[17];
  const float* b_fc    = (const float*)d_in[18];

  int n = in_sizes[0] / 128;  // number of nodes
  bf16_t* wsw  = (bf16_t*)d_ws;
  bf16_t* xw1t = wsw + OFF_TOTAL;

  convert_weights_kernel<<<(OFF_TOTAL + 255) / 256, 256, 0, stream>>>(
      w_ih_f1, w_ih_b1, w_ih_f2, w_ih_b2, w_g1, w_g2, w_fc, wsw);

  fused_gru_kernel<<<n / 128, 128, 0, stream>>>(
      x, wsw, xw1t, b_ih_f1, b_hh_f1, b_ih_b1, b_hh_b1, b_ih_f2, b_hh_f2,
      b_ih_b2, b_hh_b2, n);

  fused_gcn_kernel<<<n / 64, 128, 0, stream>>>(
      wsw, xw1t, b_g1, b_g2, b_fc, (float*)d_out, n);
}
